// Model_62886911148226
// MI455X (gfx1250) — compile-verified, matrix operand order checked by reading, not run on hardware
//
#include <hip/hip_runtime.h>

// ---------------------------------------------------------------------------
// MI455X (gfx1250) implementation of the Mamba time-series forecaster.
// All dense layers run through v_wmma_f32_16x16x32_f16 (f16 in, f32 acc).
// wave32 / WGP model: 256-thread blocks = 8 waves, each wave owns a 16x16 tile.
// ---------------------------------------------------------------------------

#define DEV __device__ __forceinline__

typedef __attribute__((ext_vector_type(16))) _Float16 half16;
typedef __attribute__((ext_vector_type(8)))  _Float16 half8;
typedef __attribute__((ext_vector_type(8)))  float    float8;

DEV float sigmoid_f(float x) { return 1.0f / (1.0f + __expf(-x)); }
DEV float silu_f(float x)    { return x * sigmoid_f(x); }
DEV float gelu_f(float x)    { return 0.5f * x * (1.0f + erff(x * 0.70710678118f)); }
DEV float softplus_f(float x){ return (x > 20.0f) ? x : log1pf(__expf(x)); }

template<int ACT> DEV float act_f(float x) {
  if (ACT == 1) return gelu_f(x);
  if (ACT == 2) return silu_f(x);
  if (ACT == 3) return sigmoid_f(x);
  if (ACT == 4) return softplus_f(x);
  return x;
}

DEV unsigned pack2h(float a, float b) {
  unsigned short lo = __builtin_bit_cast(unsigned short, (_Float16)a);
  unsigned short hi = __builtin_bit_cast(unsigned short, (_Float16)b);
  return (unsigned)lo | ((unsigned)hi << 16);
}

// fragment build + wmma: element i has K=(i&7)+16*(i>>3)+8*half, i.e. two
// contiguous 16-byte runs along K per lane -> 2x ds_load_b128 each operand.
DEV void frag_mma(const unsigned (&As)[64][20], const unsigned (&Bt)[32][20],
                  int wm, int wn, int r16, int half, float8& acc)
{
  const half8* pa = (const half8*)&As[wm + r16][half << 2];
  const half8* pb = (const half8*)&Bt[wn + r16][half << 2];
  half8 a0 = pa[0], a1 = *(const half8*)((const char*)pa + 32);
  half8 b0 = pb[0], b1 = *(const half8*)((const char*)pb + 32);
  half16 af = __builtin_shufflevector(a0, a1, 0, 1, 2, 3, 4, 5, 6, 7,
                                      8, 9, 10, 11, 12, 13, 14, 15);
  half16 bf = __builtin_shufflevector(b0, b1, 0, 1, 2, 3, 4, 5, 6, 7,
                                      8, 9, 10, 11, 12, 13, 14, 15);
  acc = __builtin_amdgcn_wmma_f32_16x16x32_f16(
      false, af, false, bf, (short)0, acc, false, false);
}

// ---------------------------------------------------------------------------
// WMMA GEMM: C[M,N] = act(A[M,K](lda) @ B[K,N] + bias)
// Macro-tile 64(M) x 32(N), K-step 32, 8 waves arranged 4(M) x 2(N).
// Fast path (interior blocks, K%32==0, even lda): unconditional b64/b32 loads.
// Guarded path: clamped addresses * {0,1} mask so loads stay unpredicated.
// ---------------------------------------------------------------------------
template<int ACT>
__global__ __launch_bounds__(256)
void gemm_wmma(const float* __restrict__ A, int lda,
               const float* __restrict__ B,
               const float* __restrict__ bias,
               float* __restrict__ C, int M, int N, int K)
{
  // rows of 40 halves (80 B): keeps every 16-byte fragment run aligned.
  __shared__ __align__(16) unsigned As[64][20];  // [m][k/2] packed f16 pairs
  __shared__ __align__(16) unsigned Bt[32][20];  // [n][k/2] packed f16 pairs

  const int tid   = threadIdx.x;
  const int tileM = blockIdx.y * 64;
  const int tileN = blockIdx.x * 32;
  const int lane  = tid & 31;
  const int wave  = tid >> 5;
  const int wm    = (wave & 3) << 4;   // 0,16,32,48
  const int wn    = (wave >> 2) << 4;  // 0,16
  const int half  = lane >> 4;
  const int r16   = lane & 15;

  float8 acc = {};

  const bool fast = (tileM + 64 <= M) && (tileN + 32 <= N) &&
                    ((K & 31) == 0) && ((lda & 1) == 0);

  if (fast) {
    for (int kk = 0; kk < K; kk += 32) {
      // A tile 64x32: 4 b64 loads per thread, no bounds logic
#pragma unroll
      for (int e = 0; e < 4; ++e) {
        int p = e * 256 + tid;           // 0..1023 float2 chunk id
        int r = p >> 4, c2 = p & 15;     // row, chunk-of-2 along K
        float2 v = *(const float2*)(A + (size_t)(tileM + r) * lda + kk + 2 * c2);
        As[r][c2] = pack2h(v.x, v.y);
      }
      // B tile 32x32 transposed: 2 packed pairs per thread, coalesced over N
#pragma unroll
      for (int e = 0; e < 2; ++e) {
        int p = e * 256 + tid;           // 0..511 pair id
        int n = p & 31, kh = (p >> 5) << 1;
        const float* pb = B + (size_t)(kk + kh) * N + tileN + n;
        Bt[n][kh >> 1] = pack2h(pb[0], pb[N]);
      }
      if (kk + 32 < K) {
        int pm = tileM + (tid & 63);
        __builtin_prefetch(&A[(size_t)pm * lda + kk + 32], 0, 1);
      }
      __syncthreads();
      frag_mma(As, Bt, wm, wn, r16, half, acc);
      __syncthreads();
    }
  } else {
    for (int kk = 0; kk < K; kk += 32) {
#pragma unroll
      for (int e = 0; e < 4; ++e) {
        int p  = e * 256 + tid;
        int r  = p >> 4;
        int kh = (p & 15) << 1;
        int gm = tileM + r;
        int gk0 = kk + kh, gk1 = gk0 + 1;
        size_t rowoff = (size_t)(gm < M ? gm : (M - 1)) * lda;
        float m0 = (gm < M && gk0 < K) ? 1.0f : 0.0f;
        float m1 = (gm < M && gk1 < K) ? 1.0f : 0.0f;
        float v0 = A[rowoff + (gk0 < K ? gk0 : 0)] * m0;  // mask-mult keeps
        float v1 = A[rowoff + (gk1 < K ? gk1 : 0)] * m1;  // loads unpredicated
        As[r][kh >> 1] = pack2h(v0, v1);
      }
#pragma unroll
      for (int e = 0; e < 2; ++e) {
        int p  = e * 256 + tid;
        int n  = p & 31;
        int kh = (p >> 5) << 1;
        int gn = tileN + n;
        int gk0 = kk + kh, gk1 = gk0 + 1;
        size_t cl = (size_t)(gn < N ? gn : 0);
        float m0 = (gn < N && gk0 < K) ? 1.0f : 0.0f;
        float m1 = (gn < N && gk1 < K) ? 1.0f : 0.0f;
        float v0 = B[(size_t)(gk0 < K ? gk0 : 0) * N + cl] * m0;
        float v1 = B[(size_t)(gk1 < K ? gk1 : 0) * N + cl] * m1;
        Bt[n][kh >> 1] = pack2h(v0, v1);
      }
      __syncthreads();
      frag_mma(As, Bt, wm, wn, r16, half, acc);
      __syncthreads();
    }
  }

  const int n = tileN + wn + r16;
  if (fast) {
    const float bv = (bias != nullptr) ? bias[n] : 0.0f;
#pragma unroll
    for (int v = 0; v < 8; ++v) {
      int m = tileM + wm + v + (half << 3);
      C[(size_t)m * N + n] = act_f<ACT>(acc[v] + bv);
    }
  } else {
    const float bv = (bias != nullptr && n < N) ? bias[n] : 0.0f;
#pragma unroll
    for (int v = 0; v < 8; ++v) {
      int m = tileM + wm + v + (half << 3);
      if (m < M && n < N)
        C[(size_t)m * N + n] = act_f<ACT>(acc[v] + bv);
    }
  }
}

// ---------------------------------------------------------------------------
// Elementwise / fused kernels
// ---------------------------------------------------------------------------
#define GID ((size_t)blockIdx.x * 256 + threadIdx.x)

// RevIN: x_enc (B,L,V) -> x_in (B*V, L) normalized; store mean/std per (b,v)
__global__ void k_revin(const float* __restrict__ x, float* __restrict__ xin,
                        float* __restrict__ meanv, float* __restrict__ stdv,
                        int Bn, int L, int V)
{
  size_t i = GID; if (i >= (size_t)Bn * V) return;
  int v = (int)(i % V), b = (int)(i / V);
  float s = 0.f;
  for (int l = 0; l < L; ++l) s += x[((size_t)b * L + l) * V + v];
  float m = s / L, q = 0.f;
  for (int l = 0; l < L; ++l) { float d = x[((size_t)b * L + l) * V + v] - m; q += d * d; }
  float sd = sqrtf(q / L + 1e-5f);
  for (int l = 0; l < L; ++l)
    xin[i * L + l] = (x[((size_t)b * L + l) * V + v] - m) / sd;
  meanv[i] = m; stdv[i] = sd;
}

__global__ void k_halve(const float* __restrict__ x, float* __restrict__ y, size_t n, int Lh)
{
  size_t i = GID; if (i >= n) return;
  int t = (int)(i % Lh); size_t r = i / Lh;
  const float* p = x + r * (size_t)(2 * Lh) + 2 * t;
  y[i] = 0.5f * (p[0] + p[1]);
}

// rf = concat([x, |rfft(x)|])  (naive DFT, Lc <= 96)
__global__ void k_dft(const float* __restrict__ x, float* __restrict__ rf, int rows, int Lc)
{
  int RD = Lc + Lc / 2 + 1;
  size_t i = GID; if (i >= (size_t)rows * RD) return;
  int c = (int)(i % RD); size_t r = i / RD;
  const float* xr = x + r * (size_t)Lc;
  if (c < Lc) { rf[i] = xr[c]; return; }
  int k = c - Lc;
  float re = 0.f, im = 0.f;
  float w = -6.283185307179586f * (float)k / (float)Lc;
  for (int t = 0; t < Lc; ++t) {
    float a = w * (float)t;
    re += xr[t] * __cosf(a);
    im += xr[t] * __sinf(a);
  }
  rf[i] = sqrtf(re * re + im * im);
}

// top-4 of 6 logits -> softmax -> scattered gates
__global__ void k_topk(const float* __restrict__ logits, float* __restrict__ gates, int rows)
{
  size_t r = GID; if (r >= (size_t)rows) return;
  float v[6]; bool used[6];
  for (int e = 0; e < 6; ++e) { v[e] = logits[r * 6 + e]; used[e] = false; }
  float tv[4]; int ti[4];
  for (int j = 0; j < 4; ++j) {
    float best = -1e30f; int bi = 0;
    for (int e = 0; e < 6; ++e)
      if (!used[e] && v[e] > best) { best = v[e]; bi = e; }
    used[bi] = true; tv[j] = best; ti[j] = bi;
  }
  float mx = tv[0], s = 0.f, ex[4];
  for (int j = 0; j < 4; ++j) { ex[j] = __expf(tv[j] - mx); s += ex[j]; }
  float g[6] = {0.f, 0.f, 0.f, 0.f, 0.f, 0.f};
  for (int j = 0; j < 4; ++j) g[ti[j]] = ex[j] / s;
  for (int e = 0; e < 6; ++e) gates[r * 6 + e] = g[e];
}

// frames for patch embedding: (rows*pn, pl) with edge zero-pad
__global__ void k_frames(const float* __restrict__ x, float* __restrict__ fr,
                         int rows, int Lc, int pl, int step, int pn)
{
  size_t i = GID; if (i >= (size_t)rows * pn * pl) return;
  int j = (int)(i % pl); size_t row = i / pl;
  int pidx = (int)(row % pn); size_t bv = row / pn;
  int t = pidx * step + j - pl / 2;
  fr[i] = (t >= 0 && t < Lc) ? x[bv * (size_t)Lc + t] : 0.0f;
}

// stacked[(b*2+s)*V + v, :] += gate[bv,e] * src[bv, :]
__global__ void k_scale_accum(const float* __restrict__ src, const float* __restrict__ gates,
                              float* __restrict__ stacked, int s, int e, int V, int D)
{
  size_t i = GID; if (i >= (size_t)2048 * D) return;
  int d = (int)(i % D); size_t bv = i / D;
  int b = (int)(bv / V), v = (int)(bv % V);
  stacked[(((size_t)b * 2 + s) * V + v) * D + d] += gates[bv * 6 + e] * src[i];
}

__global__ void k_extract_scale(const float* __restrict__ inter, float* __restrict__ xs,
                                int s, int D)
{
  size_t i = GID; if (i >= (size_t)2048 * D) return;
  int d = (int)(i % D); size_t n = i / D;
  xs[i] = inter[(n * 2 + s) * (size_t)D + d];
}

__global__ void k_flip(const float* __restrict__ x, float* __restrict__ y,
                       int Ns, int L, int d)
{
  size_t i = GID; if (i >= (size_t)Ns * L * d) return;
  int c = (int)(i % d); size_t t = i / d;
  int l = (int)(t % L); size_t n = t / L;
  y[i] = x[((n * (size_t)L) + (L - 1 - l)) * d + c];
}

// xc = silu(depthwise causal conv(xi) + b); xi = cols [0,di) of inW (T, 2di)
__global__ void k_conv_silu(const float* __restrict__ inW, const float* __restrict__ cw,
                            const float* __restrict__ cb, float* __restrict__ xc,
                            int Ns, int L, int di, int dc)
{
  size_t i = GID; if (i >= (size_t)Ns * L * di) return;
  int c = (int)(i % di); size_t tl = i / di;
  int l = (int)(tl % L); size_t n = tl / L;
  float acc = cb[c];
  for (int k = 0; k < dc; ++k) {
    int ls = l + k - (dc - 1);
    if (ls >= 0)
      acc += inW[((n * (size_t)L) + ls) * (size_t)(2 * di) + c] * cw[c * dc + k];
  }
  xc[i] = silu_f(acc);
}

// selective scan: one thread per (sequence, channel); h[16] lives in VGPRs.
// ymod = (scan_out + D*xc) * silu(z)
__global__ void k_scan(const float* __restrict__ xc, const float* __restrict__ delta,
                       const float* __restrict__ xdb, const float* __restrict__ inW,
                       const float* __restrict__ A_log, const float* __restrict__ Dp,
                       float* __restrict__ ymod, int Ns, int L, int di, int dt_rank)
{
  size_t i = GID; if (i >= (size_t)Ns * di) return;
  int c = (int)(i % di); size_t n = i / di;
  int nx = dt_rank + 32;
  float A[16], h[16];
#pragma unroll
  for (int s = 0; s < 16; ++s) { A[s] = -__expf(A_log[c * 16 + s]); h[s] = 0.f; }
  float Dc = Dp[c];
  for (int l = 0; l < L; ++l) {
    size_t t = n * (size_t)L + l;
    float dt = delta[t * di + c];
    float xv = xc[t * di + c];
    const float* bc = xdb + t * (size_t)nx + dt_rank;
    float y = 0.f;
#pragma unroll
    for (int s = 0; s < 16; ++s) {
      h[s] = __expf(dt * A[s]) * h[s] + dt * bc[s] * xv;
      y += h[s] * bc[16 + s];
    }
    float zv = inW[t * (size_t)(2 * di) + di + c];
    ymod[t * di + c] = (y + Dc * xv) * silu_f(zv);
  }
}

__global__ void k_concat2(const float* __restrict__ a, const float* __restrict__ b,
                          float* __restrict__ c, size_t T, int d)
{
  size_t i = GID; if (i >= T * (size_t)(2 * d)) return;
  int j = (int)(i % (2 * d)); size_t t = i / (2 * d);
  c[i] = (j < d) ? a[t * d + j] : b[t * d + (j - d)];
}

__global__ void k_gate_mix(const float* __restrict__ zg, const float* __restrict__ yf,
                           const float* __restrict__ yb, float* __restrict__ y, size_t n)
{
  size_t i = GID; if (i >= n) return;
  float g = zg[i];
  y[i] = g * yf[i] + (1.0f - g) * yb[i];
}

__global__ void k_add(const float* a, const float* b, float* o, size_t n)
{ size_t i = GID; if (i < n) o[i] = a[i] + b[i]; }

__global__ void k_sub(const float* a, const float* b, float* o, size_t n)
{ size_t i = GID; if (i < n) o[i] = a[i] - b[i]; }

// out = LayerNorm(a + b) per row of d
__global__ void k_add_ln(const float* __restrict__ a, const float* __restrict__ b,
                         const float* __restrict__ g, const float* __restrict__ bb,
                         float* __restrict__ out, int rows, int d)
{
  size_t r = GID; if (r >= (size_t)rows) return;
  const float* pa = a + r * (size_t)d;
  const float* pb = b + r * (size_t)d;
  float* po = out + r * (size_t)d;
  float s1 = 0.f, s2 = 0.f;
  for (int i = 0; i < d; ++i) { float t = pa[i] + pb[i]; po[i] = t; s1 += t; s2 += t * t; }
  float m = s1 / d, var = s2 / d - m * m;
  float inv = rsqrtf(var + 1e-5f);
  for (int i = 0; i < d; ++i) po[i] = (po[i] - m) * inv * g[i] + bb[i];
}

// x = LayerNorm(y1*y0 + x) in-place per row of d
__global__ void k_mul_add_ln(const float* __restrict__ y1, const float* __restrict__ y0,
                             float* __restrict__ x, const float* __restrict__ g,
                             const float* __restrict__ bb, int rows, int d)
{
  size_t r = GID; if (r >= (size_t)rows) return;
  const float* p1 = y1 + r * (size_t)d;
  const float* p0 = y0 + r * (size_t)d;
  float* px = x + r * (size_t)d;
  float s1 = 0.f, s2 = 0.f;
  for (int i = 0; i < d; ++i) { float t = p1[i] * p0[i] + px[i]; px[i] = t; s1 += t; s2 += t * t; }
  float m = s1 / d, var = s2 / d - m * m;
  float inv = rsqrtf(var + 1e-5f);
  for (int i = 0; i < d; ++i) px[i] = (px[i] - m) * inv * g[i] + bb[i];
}

// (Bn,R,C) -> (Bn,C,R)
__global__ void k_transpose(const float* __restrict__ x, float* __restrict__ y,
                            int Bn, int R, int C)
{
  size_t i = GID; if (i >= (size_t)Bn * R * C) return;
  int c = (int)(i % C); size_t t = i / C;
  int r = (int)(t % R); int b = (int)(t / R);
  y[((size_t)b * C + c) * R + r] = x[i];
}

// 13-tap moving average with edge replication, per row of len
__global__ void k_movavg(const float* __restrict__ x, float* __restrict__ t,
                         int rows, int len)
{
  size_t i = GID; if (i >= (size_t)rows * len) return;
  int j = (int)(i % len); size_t r = i / len;
  const float* xr = x + r * (size_t)len;
  float s = 0.f;
  for (int k = -6; k <= 6; ++k) {
    int jj = j + k; jj = jj < 0 ? 0 : (jj >= len ? len - 1 : jj);
    s += xr[jj];
  }
  t[i] = s * (1.0f / 13.0f);
}

// dec[b, t, v] += st[(b*V+v)*PRED + t]
__global__ void k_proj_accum(const float* __restrict__ st, float* __restrict__ dec,
                             int V, int PRED)
{
  size_t i = GID; if (i >= (size_t)2048 * PRED) return;
  int t = (int)(i % PRED); size_t bv = i / PRED;
  int b = (int)(bv / V), v = (int)(bv % V);
  dec[((size_t)b * PRED + t) * V + v] += st[i];
}

__global__ void k_final(const float* __restrict__ dec, const float* __restrict__ stdv,
                        const float* __restrict__ meanv, float* __restrict__ out,
                        int PRED, int V)
{
  size_t i = GID; if (i >= (size_t)16 * PRED * V) return;
  int v = (int)(i % V); int b = (int)(i / ((size_t)PRED * V));
  out[i] = dec[i] * stdv[b * V + v] + meanv[b * V + v];
}

// ---------------------------------------------------------------------------
// Host side: parameter unpacking (JAX pytree order: dicts sorted by key),
// workspace layout and orchestration.
// ---------------------------------------------------------------------------
struct MambaP {
  const float *A_log, *D, *W_dt, *W_in, *W_out, *W_x, *b_dt, *conv_b, *conv_w;
  int d, di, dt_rank, dconv;
};
struct BiP { MambaP b, f; const float *gate_W, *gate_b, *out_W, *out_b; };
struct EmbLayerP { const float *W1, *W2, *b1, *b2; };
struct EmbP { const float *gW1, *gW2, *gb1, *gb2; EmbLayerP lay[6]; };
struct EncP { BiP cm; const float *n1b, *n1g, *n2b, *n2g; MambaP tm; };

static MambaP read_mamba(void* const* din, int& i, int d, int dconv) {
  MambaP m; m.d = d; m.di = 2 * d; m.dt_rank = (d + 15) / 16; m.dconv = dconv;
  m.A_log  = (const float*)din[i++]; m.D      = (const float*)din[i++];
  m.W_dt   = (const float*)din[i++]; m.W_in   = (const float*)din[i++];
  m.W_out  = (const float*)din[i++]; m.W_x    = (const float*)din[i++];
  m.b_dt   = (const float*)din[i++]; m.conv_b = (const float*)din[i++];
  m.conv_w = (const float*)din[i++];
  return m;
}
static BiP read_bimamba(void* const* din, int& i, int d, int dconv) {
  BiP p;
  p.b = read_mamba(din, i, d, dconv);
  p.f = read_mamba(din, i, d, dconv);
  p.gate_W = (const float*)din[i++]; p.gate_b = (const float*)din[i++];
  p.out_W  = (const float*)din[i++]; p.out_b  = (const float*)din[i++];
  return p;
}
static EmbP read_emb(void* const* din, int& i) {
  EmbP p;
  p.gW1 = (const float*)din[i++]; p.gW2 = (const float*)din[i++];
  p.gb1 = (const float*)din[i++]; p.gb2 = (const float*)din[i++];
  for (int e = 0; e < 6; ++e) {
    p.lay[e].W1 = (const float*)din[i++]; p.lay[e].W2 = (const float*)din[i++];
    p.lay[e].b1 = (const float*)din[i++]; p.lay[e].b2 = (const float*)din[i++];
  }
  return p;
}
static EncP read_encoder(void* const* din, int& i) {
  EncP p;
  p.cm  = read_bimamba(din, i, 512, 1);
  p.n1b = (const float*)din[i++]; p.n1g = (const float*)din[i++];
  p.n2b = (const float*)din[i++]; p.n2g = (const float*)din[i++];
  p.tm  = read_mamba(din, i, 128, 4);
  return p;
}

struct WS {
  float *x_in, *x2, *meanv, *stdv, *stacked, *interbuf;
  float *rf, *gh, *logits, *gates, *frames, *hbuf, *embtmp;
  float *xs, *trendb, *seasb, *sum_st, *projtmp, *dec;
  float *enc_bm, *enc_y0, *enc_t0, *enc_t1, *enc_y1;
  float *m_inW, *m_xc, *m_delta, *m_xdb, *m_ymod;
  float *bm_xflip, *bm_yf, *bm_yb, *bm_cat, *bm_zg, *bm_ymix;
};

static inline dim3 g1(size_t n) { return dim3((unsigned)((n + 255) / 256)); }

static void gemm(hipStream_t s, const float* A, int lda, const float* B,
                 const float* bias, float* C, int M, int N, int K, int act)
{
  dim3 g((unsigned)((N + 31) / 32), (unsigned)((M + 63) / 64)), b(256);
  switch (act) {
    case 1: gemm_wmma<1><<<g, b, 0, s>>>(A, lda, B, bias, C, M, N, K); break;
    case 2: gemm_wmma<2><<<g, b, 0, s>>>(A, lda, B, bias, C, M, N, K); break;
    case 3: gemm_wmma<3><<<g, b, 0, s>>>(A, lda, B, bias, C, M, N, K); break;
    case 4: gemm_wmma<4><<<g, b, 0, s>>>(A, lda, B, bias, C, M, N, K); break;
    default: gemm_wmma<0><<<g, b, 0, s>>>(A, lda, B, bias, C, M, N, K); break;
  }
}

static void run_mamba(hipStream_t st, const MambaP& p, const float* x,
                      int Ns, int L, float* out, WS& w)
{
  const int d = p.d, di = p.di, nx = p.dt_rank + 32;
  const int T = Ns * L;
  gemm(st, x, d, p.W_in, nullptr, w.m_inW, T, 2 * di, d, 0);
  k_conv_silu<<<g1((size_t)T * di), 256, 0, st>>>(w.m_inW, p.conv_w, p.conv_b,
                                                  w.m_xc, Ns, L, di, p.dconv);
  gemm(st, w.m_xc, di, p.W_x, nullptr, w.m_xdb, T, nx, di, 0);
  gemm(st, w.m_xdb, nx, p.W_dt, p.b_dt, w.m_delta, T, di, p.dt_rank, 4);
  k_scan<<<g1((size_t)Ns * di), 256, 0, st>>>(w.m_xc, w.m_delta, w.m_xdb, w.m_inW,
                                              p.A_log, p.D, w.m_ymod, Ns, L, di, p.dt_rank);
  gemm(st, w.m_ymod, di, p.W_out, nullptr, out, T, d, di, 0);
}

static void run_bimamba(hipStream_t st, const BiP& p, const float* x,
                        int Ns, int L, float* out, WS& w)
{
  const int d = p.f.d; const size_t T = (size_t)Ns * L;
  run_mamba(st, p.f, x, Ns, L, w.bm_yf, w);
  k_flip<<<g1(T * d), 256, 0, st>>>(x, w.bm_xflip, Ns, L, d);
  run_mamba(st, p.b, w.bm_xflip, Ns, L, w.bm_ymix, w);
  k_flip<<<g1(T * d), 256, 0, st>>>(w.bm_ymix, w.bm_yb, Ns, L, d);
  k_concat2<<<g1(T * 2 * d), 256, 0, st>>>(w.bm_yf, w.bm_yb, w.bm_cat, T, d);
  gemm(st, w.bm_cat, 2 * d, p.gate_W, p.gate_b, w.bm_zg, (int)T, d, 2 * d, 3);
  k_gate_mix<<<g1(T * d), 256, 0, st>>>(w.bm_zg, w.bm_yf, w.bm_yb, w.bm_ymix, T * d);
  gemm(st, w.bm_ymix, d, p.out_W, p.out_b, out, (int)T, d, d, 0);
}

static void run_encoder(hipStream_t st, const EncP& p, float* x, WS& w)
{
  const int Bn = 16, L = 128, D = 512, rows = Bn * L;
  run_bimamba(st, p.cm, x, Bn, L, w.enc_bm, w);
  k_add_ln<<<g1(rows), 256, 0, st>>>(w.enc_bm, x, p.n1g, p.n1b, w.enc_y0, rows, D);
  k_transpose<<<g1((size_t)rows * D), 256, 0, st>>>(w.enc_y0, w.enc_t0, Bn, L, D);
  run_mamba(st, p.tm, w.enc_t0, Bn, D, w.enc_t1, w);
  k_transpose<<<g1((size_t)rows * D), 256, 0, st>>>(w.enc_t1, w.enc_y1, Bn, D, L);
  k_mul_add_ln<<<g1(rows), 256, 0, st>>>(w.enc_y1, w.enc_y0, x, p.n2g, p.n2b, rows, D);
}

static const int PATCH_H[6] = {96, 48, 24, 12, 6, 3};

static void run_emb(hipStream_t st, const EmbP& p, const float* xs, int Lc, int s, WS& w)
{
  const int rows = 2048, RD = Lc + Lc / 2 + 1;
  k_dft<<<g1((size_t)rows * RD), 256, 0, st>>>(xs, w.rf, rows, Lc);
  gemm(st, w.rf, RD, p.gW1, p.gb1, w.gh, rows, 128, RD, 1);
  gemm(st, w.gh, 128, p.gW2, p.gb2, w.logits, rows, 6, 128, 0);
  k_topk<<<g1(rows), 256, 0, st>>>(w.logits, w.gates, rows);
  for (int e = 0; e < 6; ++e) {
    int pl = PATCH_H[e], step = pl / 2, pn = Lc / step + 1, dmi = 512 / pn;
    k_frames<<<g1((size_t)rows * pn * pl), 256, 0, st>>>(xs, w.frames, rows, Lc, pl, step, pn);
    gemm(st, w.frames, pl, p.lay[e].W1, p.lay[e].b1, w.hbuf, rows * pn, dmi, pl, 1);
    gemm(st, w.hbuf, pn * dmi, p.lay[e].W2, p.lay[e].b2, w.embtmp, rows, 512, pn * dmi, 0);
    k_scale_accum<<<g1((size_t)rows * 512), 256, 0, st>>>(w.embtmp, w.gates, w.stacked,
                                                          s, e, 128, 512);
  }
}

extern "C" void kernel_launch(void* const* d_in, const int* in_sizes, int n_in,
                              void* d_out, int out_size, void* d_ws, size_t ws_size,
                              hipStream_t stream)
{
  (void)in_sizes; (void)n_in; (void)out_size; (void)ws_size;
  const float* x_enc = (const float*)d_in[0];
  // d_in[1..3]: x_mark_enc, x_dec, x_mark_dec (unused)

  // params pytree (dicts sorted by key): cross, embs, proj, seasonal, trend
  int pi = 4;
  BiP cross = read_bimamba(d_in, pi, 512, 4);
  EmbP embs[2]; embs[0] = read_emb(d_in, pi); embs[1] = read_emb(d_in, pi);
  const float* projW[2]; const float* projB[2];
  projW[0] = (const float*)d_in[pi++]; projB[0] = (const float*)d_in[pi++];
  projW[1] = (const float*)d_in[pi++]; projB[1] = (const float*)d_in[pi++];
  EncP seas[2]  = { read_encoder(d_in, pi), read_encoder(d_in, pi) };
  EncP trend[2] = { read_encoder(d_in, pi), read_encoder(d_in, pi) };

  // workspace bump allocation
  float* base = (float*)d_ws;
  size_t off = 0;
  auto alloc = [&](size_t n) -> float* {
    float* p = base + off; off += (n + 63) & ~(size_t)63; return p;
  };
  WS w;
  w.x_in     = alloc(2048 * 96);
  w.x2       = alloc(2048 * 48);
  w.meanv    = alloc(2048);
  w.stdv     = alloc(2048);
  w.stacked  = alloc(2097152);
  w.interbuf = alloc(2097152);
  w.rf       = alloc(2048 * 145);
  w.gh       = alloc(2048 * 128);
  w.logits   = alloc(2048 * 6);
  w.gates    = alloc(2048 * 6);
  w.frames   = alloc(600064);
  w.hbuf     = alloc(1048576);
  w.embtmp   = alloc(1048576);
  w.xs       = alloc(1048576);
  w.trendb   = alloc(1048576);
  w.seasb    = alloc(1048576);
  w.sum_st   = alloc(1048576);
  w.projtmp  = alloc(2048 * 96);
  w.dec      = alloc(16 * 96 * 128);
  w.enc_bm   = alloc(1048576);
  w.enc_y0   = alloc(1048576);
  w.enc_t0   = alloc(1048576);
  w.enc_t1   = alloc(1048576);
  w.enc_y1   = alloc(1048576);
  w.m_inW    = alloc(8388608);
  w.m_xc     = alloc(4194304);
  w.m_delta  = alloc(4194304);
  w.m_xdb    = alloc(524288);
  w.m_ymod   = alloc(4194304);
  w.bm_xflip = alloc(2097152);
  w.bm_yf    = alloc(2097152);
  w.bm_yb    = alloc(2097152);
  w.bm_cat   = alloc(4194304);
  w.bm_zg    = alloc(2097152);
  w.bm_ymix  = alloc(2097152);

  // --- pipeline ---
  k_revin<<<g1(2048), 256, 0, stream>>>(x_enc, w.x_in, w.meanv, w.stdv, 16, 96, 128);
  k_halve<<<g1((size_t)2048 * 48), 256, 0, stream>>>(w.x_in, w.x2, (size_t)2048 * 48, 48);

  hipMemsetAsync(w.stacked, 0, sizeof(float) * 2097152, stream);
  run_emb(stream, embs[0], w.x_in, 96, 0, w);
  run_emb(stream, embs[1], w.x2, 48, 1, w);

  // cross bimamba over (B*V=2048, S=2, D=512) flat view
  run_bimamba(stream, cross, w.stacked, 2048, 2, w.interbuf, w);

  hipMemsetAsync(w.dec, 0, sizeof(float) * 16 * 96 * 128, stream);
  for (int s = 0; s < 2; ++s) {
    k_extract_scale<<<g1((size_t)2048 * 512), 256, 0, stream>>>(w.interbuf, w.xs, s, 512);
    k_movavg<<<g1((size_t)2048 * 512), 256, 0, stream>>>(w.xs, w.trendb, 2048, 512);
    k_sub<<<g1((size_t)2048 * 512), 256, 0, stream>>>(w.xs, w.trendb, w.seasb,
                                                      (size_t)2048 * 512);
    for (int l = 0; l < 2; ++l) run_encoder(stream, seas[l], w.seasb, w);
    for (int l = 0; l < 2; ++l) run_encoder(stream, trend[l], w.trendb, w);
    k_add<<<g1((size_t)2048 * 512), 256, 0, stream>>>(w.seasb, w.trendb, w.sum_st,
                                                      (size_t)2048 * 512);
    gemm(stream, w.sum_st, 512, projW[s], projB[s], w.projtmp, 2048, 96, 512, 0);
    k_proj_accum<<<g1((size_t)2048 * 96), 256, 0, stream>>>(w.projtmp, w.dec, 128, 96);
  }
  k_final<<<g1((size_t)16 * 96 * 128), 256, 0, stream>>>(w.dec, w.stdv, w.meanv,
                                                         (float*)d_out, 96, 128);
}
